// GCN_46145128628406
// MI455X (gfx1250) — compile-verified, matrix operand order checked by reading, not run on hardware
//
#include <hip/hip_runtime.h>
#include <hip/hip_bf16.h>

#define NODES   100000     // multiple of 16 -> no tail handling in WMMA GEMM
#define EDGES   1600000
#define FEAT    128
#define NCLS    10
#define NGRAPH  64

static_assert(NODES % 16 == 0, "GEMM assumes full 16-row tiles");

typedef __attribute__((ext_vector_type(16))) __bf16 v16bf_t;
typedef __attribute__((ext_vector_type(8)))  __bf16 v8bf_t;
typedef __attribute__((ext_vector_type(8)))  float  v8f_t;

// ---------------------------------------------------------------- utilities
__global__ void k_fill(float* __restrict__ p, float v, int n) {
  int i = blockIdx.x * blockDim.x + threadIdx.x;
  if (i < n) p[i] = v;
}

__global__ void k_cvt_bf16(const float* __restrict__ src, __bf16* __restrict__ dst, int n) {
  int i = blockIdx.x * blockDim.x + threadIdx.x;
  if (i < n) dst[i] = (__bf16)src[i];
}

// in-degree over dst (self-loop contribution pre-seeded as 1.0 by k_fill)
__global__ void k_deg(const int* __restrict__ dst, float* __restrict__ deg, int e) {
  int i = blockIdx.x * blockDim.x + threadIdx.x;
  if (i < e) unsafeAtomicAdd(&deg[dst[i]], 1.0f);
}

__global__ void k_dinv(const float* __restrict__ deg, float* __restrict__ dinv, int n) {
  int i = blockIdx.x * blockDim.x + threadIdx.x;
  if (i < n) {
    float d = deg[i];
    dinv[i] = d > 0.0f ? rsqrtf(d) : 0.0f;
  }
}

// ------------------------------------------------- weight pre-pack (per layer)
// Repack W[128x128] fp32 row-major into bf16 "WMMA B-fragment order":
// flat index t = ((nTile*4 + kStep)*32 + lane)*16 + j, so each wave's B
// operand is ONE aligned 32-byte vector load per lane (2x global_load_b128).
// 32 KB total -> L0/L2 resident.
__global__ void k_packW(const float* __restrict__ W, __bf16* __restrict__ Wp) {
  int t = blockIdx.x * blockDim.x + threadIdx.x;
  if (t >= FEAT * FEAT) return;
  int j    = t & 15;
  int lane = (t >> 4) & 31;
  int ks   = (t >> 9) & 3;
  int nt   = t >> 11;                 // 0..7
  int half = lane >> 4;
  int r    = lane & 15;
  int n = nt * 16 + r;
  int k = ks * 32 + ((j < 8) ? (half * 8 + j) : (16 + half * 8 + (j - 8)));
  Wp[t] = (__bf16)W[k * FEAT + n];
}

// ---------------------------------------------------------------- WMMA GEMM
// C[NODES x 128] = A[NODES x 128](bf16) * W[128 x 128](packed bf16), f32 accum.
// One wave32 per 16x16 tile; 8 waves/block cover all 8 column tiles of a
// 16-row strip. Hot loop per K-step: 2x b128 (A) + 2x b128 (B) + 1 WMMA;
// epilogue: 8 unguarded global_store_b32 with immediate offsets.
__global__ void k_gemm_wmma(const __bf16* __restrict__ A,
                            const __bf16* __restrict__ Wp,
                            float* __restrict__ C) {
  const int wave = threadIdx.x >> 5;   // 0..7 -> column tile
  const int lane = threadIdx.x & 31;
  const int half = lane >> 4;          // 0: lanes 0-15, 1: lanes 16-31
  const int r    = lane & 15;

  const int mBase = blockIdx.x * 16;

  const __bf16* __restrict__ arow = A + (size_t)(mBase + r) * FEAT;
  const v16bf_t* __restrict__ bfrag =
      reinterpret_cast<const v16bf_t*>(Wp) + ((size_t)wave * 4 * 32 + lane);

  v8f_t acc = {};
#pragma unroll
  for (int ks = 0; ks < 4; ++ks) {
    const int kb = ks * 32;
    // A 16x32 bf16 fragment (ISA 05_wmma layout): lanes 0-15 hold
    // K {0..7,16..23}, lanes 16-31 hold K {8..15,24..31}. Two aligned
    // 16B loads, combined by pure register shuffles.
    v8bf_t alo = *reinterpret_cast<const v8bf_t*>(arow + kb + half * 8);
    v8bf_t ahi = *reinterpret_cast<const v8bf_t*>(arow + kb + 16 + half * 8);
    v16bf_t a = __builtin_shufflevector(alo, ahi, 0, 1, 2, 3, 4, 5, 6, 7,
                                        8, 9, 10, 11, 12, 13, 14, 15);
    // B fragment: single 32B vector load from the pre-packed image.
    v16bf_t b = bfrag[ks * 32];
    acc = __builtin_amdgcn_wmma_f32_16x16x32_bf16(
        /*neg_a=*/false, a, /*neg_b=*/false, b,
        /*c_mod=*/(short)0, acc, /*reuse_a=*/false, /*reuse_b=*/false);
  }

  // C/D layout: VGPR v holds row (mBase + half*8 + v), col = wave*16 + r.
  // NODES % 16 == 0 -> no guards; one base address + immediate offsets.
  float* __restrict__ crow =
      C + (size_t)(mBase + half * 8) * FEAT + (wave * 16 + r);
#pragma unroll
  for (int v = 0; v < 8; ++v) crow[v * FEAT] = acc[v];
}

// ------------------------------------------------------- edge gather/scatter
// One wave per edge (32 lanes x float4 = 128 feats); coalesced 512B node-row
// gathers and L2-resident global_atomic_add_f32 (working set < 192MB L2).
__global__ void k_scatter(const float* __restrict__ HW,
                          const int* __restrict__ src,
                          const int* __restrict__ dst,
                          const float* __restrict__ dinv,
                          float* __restrict__ agg) {
  int idx = blockIdx.x * blockDim.x + threadIdx.x;
  if (idx >= EDGES * 32) return;
  int e = idx >> 5;
  int f = (idx & 31) << 2;
  int s = src[e];
  int d = dst[e];
  float nrm = dinv[s] * dinv[d];
  const float4 v = *reinterpret_cast<const float4*>(HW + (size_t)s * FEAT + f);
  float* ap = agg + (size_t)d * FEAT + f;
  unsafeAtomicAdd(ap + 0, v.x * nrm);
  unsafeAtomicAdd(ap + 1, v.y * nrm);
  unsafeAtomicAdd(ap + 2, v.z * nrm);
  unsafeAtomicAdd(ap + 3, v.w * nrm);
}

// epilogue: add self-loop term dinv^2 * HW, bias, ReLU; emit bf16 activations
// (next layer's WMMA A-operand / final pooling input). AGG is dead after this.
__global__ void k_finalize(const float* __restrict__ agg,
                           const float* __restrict__ HW,
                           const float* __restrict__ dinv,
                           const float* __restrict__ bias,
                           __bf16* __restrict__ Hout) {
  int idx = blockIdx.x * blockDim.x + threadIdx.x;
  if (idx >= NODES * FEAT) return;
  int i = idx >> 7;
  int f = idx & (FEAT - 1);
  float di = dinv[i];
  float v = agg[idx] + HW[idx] * di * di + bias[f];
  Hout[idx] = (__bf16)fmaxf(v, 0.0f);
}

// ---------------------------------------------------------------- pooling
__global__ void k_pool(const __bf16* __restrict__ H,
                       const int* __restrict__ batch,
                       float* __restrict__ pooled,
                       float* __restrict__ cnt) {
  int idx = blockIdx.x * blockDim.x + threadIdx.x;
  if (idx >= NODES * FEAT) return;
  int i = idx >> 7;
  int f = idx & (FEAT - 1);
  int g = batch[i];
  unsafeAtomicAdd(&pooled[(size_t)g * FEAT + f], (float)H[idx]);
  if (f == 0) unsafeAtomicAdd(&cnt[g], 1.0f);
}

__global__ void k_head(const float* __restrict__ pooled,
                       const float* __restrict__ cnt,
                       const float* __restrict__ Wl,
                       const float* __restrict__ bl,
                       float* __restrict__ out) {
  int idx = blockIdx.x * blockDim.x + threadIdx.x;
  if (idx >= NGRAPH * NCLS) return;
  int g = idx / NCLS;
  int c = idx % NCLS;
  float s = 0.0f;
#pragma unroll 16
  for (int k = 0; k < FEAT; ++k) s += pooled[(size_t)g * FEAT + k] * Wl[k * NCLS + c];
  out[idx] = s / fmaxf(cnt[g], 1.0f) + bl[c];
}

// ---------------------------------------------------------------- launch
extern "C" void kernel_launch(void* const* d_in, const int* in_sizes, int n_in,
                              void* d_out, int out_size, void* d_ws, size_t ws_size,
                              hipStream_t stream) {
  (void)in_sizes; (void)n_in; (void)out_size; (void)ws_size;

  const float* x   = (const float*)d_in[0];
  const int*   ei  = (const int*)d_in[1];       // [2, E] flattened (harness int contract)
  const int*   bat = (const int*)d_in[2];
  const float* W1  = (const float*)d_in[3];
  const float* b1  = (const float*)d_in[4];
  const float* W2  = (const float*)d_in[5];
  const float* b2  = (const float*)d_in[6];
  const float* W3  = (const float*)d_in[7];
  const float* b3  = (const float*)d_in[8];
  const float* Wl  = (const float*)d_in[9];
  const float* bl  = (const float*)d_in[10];
  float* out = (float*)d_out;

  const int* src = ei;
  const int* dst = ei + EDGES;

  // workspace carve-out (256B aligned); total ~128 MB (< 192 MB global L2)
  char* ws = (char*)d_ws;
  size_t off = 0;
  auto carve = [&](size_t bytes) -> void* {
    off = (off + 255) & ~(size_t)255;
    void* p = ws + off;
    off += bytes;
    return p;
  };
  float*  deg    = (float*)carve((size_t)NODES * 4);
  float*  dinv   = (float*)carve((size_t)NODES * 4);
  __bf16* Wp     = (__bf16*)carve((size_t)FEAT * FEAT * 2);
  __bf16* Hbf    = (__bf16*)carve((size_t)NODES * FEAT * 2);   // bf16 activations
  float*  HW     = (float*)carve((size_t)NODES * FEAT * 4);    // GEMM output
  float*  AGG    = (float*)carve((size_t)NODES * FEAT * 4);    // fp32 atomic accum
  float*  pooled = (float*)carve((size_t)NGRAPH * FEAT * 4);
  float*  cnt    = (float*)carve((size_t)NGRAPH * 4);

  const int B = 256;
  auto nblk = [](long long n) { return (unsigned)((n + 255) / 256); };

  const unsigned gemmBlocks = NODES / 16;                 // 6250
  const unsigned nfBlocks   = nblk((long long)NODES * FEAT);
  const unsigned scBlocks   = nblk((long long)EDGES * 32);

  // degrees + symmetric-norm coefficients (self loop pre-seeded = 1.0)
  k_fill<<<nblk(NODES), B, 0, stream>>>(deg, 1.0f, NODES);
  k_deg<<<nblk(EDGES), B, 0, stream>>>(dst, deg, EDGES);
  k_dinv<<<nblk(NODES), B, 0, stream>>>(deg, dinv, NODES);

  // layer-1 input -> bf16 once
  k_cvt_bf16<<<nfBlocks, B, 0, stream>>>(x, Hbf, NODES * FEAT);

  auto layer = [&](const float* Wm, const float* bias) {
    k_packW<<<nblk(FEAT * FEAT), B, 0, stream>>>(Wm, Wp);
    k_gemm_wmma<<<gemmBlocks, B, 0, stream>>>(Hbf, Wp, HW);          // consumes Hbf
    k_fill<<<nfBlocks, B, 0, stream>>>(AGG, 0.0f, NODES * FEAT);
    k_scatter<<<scBlocks, B, 0, stream>>>(HW, src, dst, dinv, AGG);
    k_finalize<<<nfBlocks, B, 0, stream>>>(AGG, HW, dinv, bias, Hbf); // rewrites Hbf
  };

  layer(W1, b1);
  layer(W2, b2);
  layer(W3, b3);

  k_fill<<<nblk(NGRAPH * FEAT), B, 0, stream>>>(pooled, 0.0f, NGRAPH * FEAT);
  k_fill<<<1, B, 0, stream>>>(cnt, 0.0f, NGRAPH);
  k_pool<<<nfBlocks, B, 0, stream>>>(Hbf, bat, pooled, cnt);
  k_head<<<nblk(NGRAPH * NCLS), B, 0, stream>>>(pooled, cnt, Wl, bl, out);
}
